// DANet_46909632807049
// MI455X (gfx1250) — compile-verified, matrix operand order checked by reading, not run on hardware
//
#include <hip/hip_runtime.h>
#include <hip/hip_bf16.h>
#include <math.h>

// ---------------------------------------------------------------------------
// DANet forward: 3x (z = A @ W + b ; act = tanh(z)), out = stack(z0,z1,z2,a0,a1,a2)
// B = 8192 rows, D = 2048. bf16 WMMA (v_wmma_f32_16x16x32_bf16), f32 accumulate.
// Global->LDS via CDNA5 async copies (ASYNCcnt), double-buffered.
// ---------------------------------------------------------------------------

typedef __attribute__((ext_vector_type(16))) __bf16 v16bf;
typedef __attribute__((ext_vector_type(8)))  float  v8f;
typedef __attribute__((ext_vector_type(4)))  __bf16 v4bf;

#define ROWS_B   8192
#define DIM_D    2048
#define TILE_M   128
#define TILE_N   128
#define TILE_K   32
#define LDS_STR  40            // bf16 elems per LDS row: 32 + 8 pad (80 B, conflict-free)

union FragU { uint4 u[2]; v16bf v; };

// ---- fp32 -> bf16 elementwise (x conversion) ------------------------------
__global__ void cvt_bf16_kernel(const float* __restrict__ src,
                                __bf16* __restrict__ dst, size_t n4) {
    size_t i = (size_t)blockIdx.x * blockDim.x + threadIdx.x;
    if (i >= n4) return;
    float4 f = ((const float4*)src)[i];
    v4bf o;
    o.x = (__bf16)f.x; o.y = (__bf16)f.y; o.z = (__bf16)f.z; o.w = (__bf16)f.w;
    ((v4bf*)dst)[i] = o;
}

// ---- fp32 -> bf16 transpose (weights: W[k][n] -> Wt[n][k]) ----------------
__global__ void cvt_transpose_kernel(const float* __restrict__ W,
                                     __bf16* __restrict__ Wt, int n) {
    __shared__ float tile[32][33];
    const int tx = threadIdx.x & 31;
    const int ty = threadIdx.x >> 5;            // 0..7
    const int bx = blockIdx.x * 32;             // col tile of W
    const int by = blockIdx.y * 32;             // row tile of W
#pragma unroll
    for (int i = 0; i < 4; ++i)
        tile[ty + i * 8][tx] = W[(size_t)(by + ty + i * 8) * n + bx + tx];
    __syncthreads();
#pragma unroll
    for (int i = 0; i < 4; ++i)
        Wt[(size_t)(bx + ty + i * 8) * n + by + tx] = (__bf16)tile[tx][ty + i * 8];
}

// branch-free tanh: 1 - 2/(e^{2z}+1); saturates to +/-1 via exp under/overflow.
// v_mul + v_exp_f32 + v_add + v_rcp_f32 + v_fma — no divergence, no div sequence.
__device__ __forceinline__ float fast_tanh(float z) {
    float t = __expf(2.0f * z);
    float r = __builtin_amdgcn_rcpf(t + 1.0f);
    return 1.0f - 2.0f * r;
}

// ---- fused GEMM + bias + tanh layer ---------------------------------------
// A:  [M x K] bf16 row-major          (activations, K contiguous)
// Bt: [N x K] bf16 row-major          (transposed weights, K contiguous)
// zout/aout: f32 [M x N], abf: bf16 [M x N] (next layer's A; may be null)
__global__ __launch_bounds__(256, 2)
void gemm_bias_tanh_kernel(const __bf16* __restrict__ A,
                           const __bf16* __restrict__ Bt,
                           const float*  __restrict__ bias,
                           float* __restrict__ zout,
                           float* __restrict__ aout,
                           __bf16* __restrict__ abf,
                           int M, int N, int K) {
    __shared__ __bf16 sA[2][TILE_M * LDS_STR];
    __shared__ __bf16 sB[2][TILE_N * LDS_STR];

    const int tid   = threadIdx.x;
    const int lane  = tid & 31;
    const int wave  = tid >> 5;        // 0..7
    const int waveM = wave & 3;        // 4 waves across M
    const int waveN = wave >> 2;       // 2 waves across N
    const int lrow  = lane & 15;
    const int khalf = lane >> 4;

    const int blockN = blockIdx.x * TILE_N;
    const int blockM = blockIdx.y * TILE_M;

    // staging assignment: 512 x 16B chunks per (A|B) tile, 2 per thread each
    int srow[2], schk[2];
#pragma unroll
    for (int i = 0; i < 2; ++i) {
        int idx = i * 256 + tid;
        srow[i] = idx >> 2;            // 0..127
        schk[i] = idx & 3;             // 8-bf16 (16 B) chunk
    }

    // this thread's global source chunks (advance by k0 elements per tile)
    const __bf16* gA[2];
    const __bf16* gB[2];
#pragma unroll
    for (int i = 0; i < 2; ++i) {
        gA[i] = A  + (size_t)(blockM + srow[i]) * K + schk[i] * 8;
        gB[i] = Bt + (size_t)(blockN + srow[i]) * K + schk[i] * 8;
    }
    // this thread's LDS destination byte offsets (addrspace(3) offset = low 32 bits)
    unsigned ldsA[2][2], ldsB[2][2];
#pragma unroll
    for (int b = 0; b < 2; ++b)
#pragma unroll
        for (int i = 0; i < 2; ++i) {
            ldsA[b][i] = (unsigned)(uintptr_t)&sA[b][srow[i] * LDS_STR + schk[i] * 8];
            ldsB[b][i] = (unsigned)(uintptr_t)&sB[b][srow[i] * LDS_STR + schk[i] * 8];
        }

    // async HBM -> LDS copy of one K-tile into buffer `buf` (no VGPR staging)
    auto issue_tile = [&](int buf, int k0) {
#pragma unroll
        for (int i = 0; i < 2; ++i) {
            asm volatile("global_load_async_to_lds_b128 %0, %1, off"
                         :: "v"(ldsA[buf][i]), "v"(gA[i] + k0) : "memory");
            asm volatile("global_load_async_to_lds_b128 %0, %1, off"
                         :: "v"(ldsB[buf][i]), "v"(gB[i] + k0) : "memory");
        }
    };

    v8f acc[2][4];
#pragma unroll
    for (int fm = 0; fm < 2; ++fm)
#pragma unroll
        for (int fn = 0; fn < 4; ++fn)
            acc[fm][fn] = (v8f)0.f;

    const int kTiles = K / TILE_K;     // 64

    // prologue: tile 0 into buf 0
    issue_tile(0, 0);
    asm volatile("s_wait_asynccnt 0" ::: "memory");
    __syncthreads();

#pragma unroll 1
    for (int kt = 0; kt < kTiles; ++kt) {
        const int cbuf = kt & 1;
        const bool more = (kt + 1) < kTiles;

        // overlap: fetch next tile straight into the other LDS buffer
        if (more) issue_tile(cbuf ^ 1, (kt + 1) * TILE_K);

        // load fragments from LDS (ISA 16-bit A/B layout: two b128 chunks/lane)
        FragU af[2], bfr[4];
#pragma unroll
        for (int fm = 0; fm < 2; ++fm) {
            const __bf16* p = &sA[cbuf][(waveM * 32 + fm * 16 + lrow) * LDS_STR + khalf * 8];
            af[fm].u[0] = *(const uint4*)p;         // K =      khalf*8 .. +7
            af[fm].u[1] = *(const uint4*)(p + 16);  // K = 16 + khalf*8 .. +7
        }
#pragma unroll
        for (int fn = 0; fn < 4; ++fn) {
            const __bf16* p = &sB[cbuf][(waveN * 64 + fn * 16 + lrow) * LDS_STR + khalf * 8];
            bfr[fn].u[0] = *(const uint4*)p;
            bfr[fn].u[1] = *(const uint4*)(p + 16);
        }

#pragma unroll
        for (int fm = 0; fm < 2; ++fm)
#pragma unroll
            for (int fn = 0; fn < 4; ++fn)
                acc[fm][fn] = __builtin_amdgcn_wmma_f32_16x16x32_bf16(
                    false, af[fm].v, false, bfr[fn].v,
                    (short)0, acc[fm][fn], false, false);

        if (more) {
            asm volatile("s_wait_asynccnt 0" ::: "memory");
            __syncthreads();
        }
    }

    // epilogue: bias + branch-free tanh; stream f32 outputs, bf16 act for next layer
    float bv[4];
#pragma unroll
    for (int fn = 0; fn < 4; ++fn)
        bv[fn] = bias[blockN + waveN * 64 + fn * 16 + lrow];

#pragma unroll
    for (int fm = 0; fm < 2; ++fm)
#pragma unroll
        for (int fn = 0; fn < 4; ++fn) {
            const int col = blockN + waveN * 64 + fn * 16 + lrow;
            v8f c = acc[fm][fn];
#pragma unroll
            for (int i = 0; i < 8; ++i) {
                const int rowg = blockM + waveM * 32 + fm * 16 + khalf * 8 + i;
                const size_t o = (size_t)rowg * N + col;
                const float z = c[i] + bv[fn];
                const float a = fast_tanh(z);
                __builtin_nontemporal_store(z, zout + o);
                __builtin_nontemporal_store(a, aout + o);
                if (abf) abf[o] = (__bf16)a;
            }
        }
}

// ---------------------------------------------------------------------------
extern "C" void kernel_launch(void* const* d_in, const int* in_sizes, int n_in,
                              void* d_out, int out_size, void* d_ws, size_t ws_size,
                              hipStream_t stream) {
    const float* x  = (const float*)d_in[0];
    const float* W0 = (const float*)d_in[1];
    const float* b0 = (const float*)d_in[2];
    const float* W1 = (const float*)d_in[3];
    const float* b1 = (const float*)d_in[4];
    const float* W2 = (const float*)d_in[5];
    const float* b2 = (const float*)d_in[6];
    float* out = (float*)d_out;

    const size_t BD = (size_t)ROWS_B * DIM_D;       // 16,777,216
    const size_t DD = (size_t)DIM_D * DIM_D;        //  4,194,304

    // workspace layout (88 MB total)
    char* ws = (char*)d_ws;
    __bf16* actA = (__bf16*)ws;                     // 32 MB
    __bf16* actB = (__bf16*)(ws + BD * 2);          // 32 MB
    __bf16* W0t  = (__bf16*)(ws + BD * 4);          //  8 MB each
    __bf16* W1t  = W0t + DD;
    __bf16* W2t  = W1t + DD;

    // x -> bf16
    {
        size_t n4 = BD / 4;
        cvt_bf16_kernel<<<(unsigned)((n4 + 255) / 256), 256, 0, stream>>>(x, actA, n4);
    }
    // weights -> bf16 transposed ([K,N] -> [N,K])
    {
        dim3 g(DIM_D / 32, DIM_D / 32);
        cvt_transpose_kernel<<<g, 256, 0, stream>>>(W0, W0t, DIM_D);
        cvt_transpose_kernel<<<g, 256, 0, stream>>>(W1, W1t, DIM_D);
        cvt_transpose_kernel<<<g, 256, 0, stream>>>(W2, W2t, DIM_D);
    }

    dim3 grid(DIM_D / TILE_N, ROWS_B / TILE_M);     // 16 x 64
    // layer 0: z0 -> out[0], a0 -> out[3], next act -> actB
    gemm_bias_tanh_kernel<<<grid, 256, 0, stream>>>(actA, W0t, b0,
        out + 0 * BD, out + 3 * BD, actB, ROWS_B, DIM_D, DIM_D);
    // layer 1: z1 -> out[1], a1 -> out[4], next act -> actA
    gemm_bias_tanh_kernel<<<grid, 256, 0, stream>>>(actB, W1t, b1,
        out + 1 * BD, out + 4 * BD, actA, ROWS_B, DIM_D, DIM_D);
    // layer 2: z2 -> out[2], a2 -> out[5], no downstream consumer
    gemm_bias_tanh_kernel<<<grid, 256, 0, stream>>>(actA, W2t, b2,
        out + 2 * BD, out + 5 * BD, (__bf16*)nullptr, ROWS_B, DIM_D, DIM_D);
}
